// TransformerModel_17480516894762
// MI455X (gfx1250) — matrix operand order repaired and checked by reading, not compiled, not run on hardware
//
#include <hip/hip_runtime.h>
#include <hip/hip_bf16.h>
#include <cstdint>
#include <cstddef>

// Problem dims (fixed by the reference)
#define B_   2
#define S_   2048
#define L_   6
#define D_   512
#define H_   8
#define F_   2048
#define V_   388
#define DK_  64
#define NTOK (B_ * S_)

typedef __attribute__((ext_vector_type(16))) __bf16       bf16x16;
typedef __attribute__((ext_vector_type(8)))  float        f32x8;
typedef __attribute__((ext_vector_type(4)))  unsigned int u32x4;
typedef __attribute__((ext_vector_type(8)))  int          i32x8;
typedef __attribute__((ext_vector_type(4)))  int          i32x4;

// ---------------------------------------------------------------------------
// CDNA5 WMMA fragment layouts (ISA 7.12.2, wave32)
//  A (bf16 16x32): lane m = lane&15, half = lane>>4
//    vgpr i : K = half*16 + 2i (contiguous k order used by this code, not the ISA table order)
//  B (bf16 32x16): lane n = lane&15; vgpr i : K = half*16 + 2i (same contiguous order as A, chosen here)
//  C/D (f32 16x16): lane n = lane&15; vgpr r : M = r + 8*half
// ---------------------------------------------------------------------------
static __device__ __forceinline__ int a_koff(int i, int half) {
  return half * 16 + 2 * i;
}

static __device__ __forceinline__ f32x8 wmma_bf16(bf16x16 a, bf16x16 b, f32x8 c) {
  return __builtin_amdgcn_wmma_f32_16x16x32_bf16(false, a, false, b, (short)0, c,
                                                 false, false);
}

// ---------------------------------------------------------------------------
// Tensor Data Mover: 2-D tile (tile_rows x tile_k bf16) global -> LDS.
// data_size=2B; pad_enable: +1 dword after every 16 dwords => LDS row stride
// = tile_k + 2*(tile_k/32) bf16 (68 for tile_k=64). OOB rows/cols read zero.
// Descriptor packing per ISA ch.8 (D# groups 0/1; 2-D so groups 2/3/4 zero).
// This toolchain exposes the 6-arg builtin: (g0, g1, g2, g3, g4, cpol).
// ---------------------------------------------------------------------------
static __device__ __forceinline__ void tdm_load_2d(unsigned lds_addr,
                                                   const void* gptr,
                                                   unsigned tile_k,
                                                   unsigned tile_rows,
                                                   unsigned rem_k,
                                                   unsigned rem_rows,
                                                   unsigned stride_elems) {
  unsigned long long ga = (unsigned long long)(size_t)gptr;
  u32x4 g0;
  g0[0] = 1u;                                   // count=1, user descriptor
  g0[1] = lds_addr;                             // lds_addr (bits 63:32)
  g0[2] = (unsigned)ga;                         // global_addr[31:0]
  g0[3] = (unsigned)((ga >> 32) & 0x01ffffffu)  // global_addr[56:32]
          | (2u << 30);                         // type=2 (image)
  i32x8 g1;
  g1[0] = (int)((1u << 16)    // data_size = 2 bytes
                | (1u << 20)  // pad_enable
                | (3u << 22)  // pad_interval: every 16 dwords
                | (0u << 25));// pad_amount: 1 dword
  g1[1] = (int)((rem_k & 0xffffu) << 16);                        // tensor_dim0 lo
  g1[2] = (int)((rem_k >> 16) | ((rem_rows & 0xffffu) << 16));   // dim0 hi | dim1 lo
  g1[3] = (int)((rem_rows >> 16) | (tile_k << 16));              // dim1 hi | tile_dim0
  g1[4] = (int)tile_rows;                                        // tile_dim1 (dim2=0)
  g1[5] = (int)stride_elems;                                     // dim0_stride[31:0]
  g1[6] = 0;
  g1[7] = 0;
  i32x4 z4 = {0, 0, 0, 0};
  i32x8 z8 = {0, 0, 0, 0, 0, 0, 0, 0};
  __builtin_amdgcn_tensor_load_to_lds(g0, g1, z4, z4, z8, 0);
}

// Fragments out of a TDM-padded LDS tile (row stride 68, mid-row pad at kk=32)
static __device__ __forceinline__ bf16x16 lds_frag68_a(const __bf16* P, int row,
                                                       int kk0, int half) {
  const __bf16* rp = &P[row * 68 + kk0 + ((kk0 >> 5) << 1)];
  union { bf16x16 v; unsigned int u[8]; } r;
#pragma unroll
  for (int i = 0; i < 8; ++i) r.u[i] = *(const unsigned int*)&rp[a_koff(i, half)];
  return r.v;
}
static __device__ __forceinline__ bf16x16 lds_frag68_b(const __bf16* P, int row,
                                                       int kk0, int half) {
  const __bf16* rp = &P[row * 68 + kk0 + ((kk0 >> 5) << 1)];
  union { bf16x16 v; unsigned int u[8]; } r;
#pragma unroll
  for (int i = 0; i < 8; ++i) r.u[i] = *(const unsigned int*)&rp[half * 16 + 2 * i];
  return r.v;
}

// Packed-dword fragment loads from bf16 global rows (stride D_)
static __device__ __forceinline__ bf16x16 afrag_gb(const __bf16* base, int m,
                                                   int kk0, int half) {
  const __bf16* r = base + (size_t)m * D_ + kk0;
  union { bf16x16 v; unsigned int u[8]; } a;
#pragma unroll
  for (int i = 0; i < 8; ++i) a.u[i] = *(const unsigned int*)&r[a_koff(i, half)];
  return a.v;
}
static __device__ __forceinline__ bf16x16 btfrag_gb(const __bf16* base, int n,
                                                    int kk0, int half) {
  const __bf16* r = base + (size_t)n * D_ + kk0;
  union { bf16x16 v; unsigned int u[8]; } b;
#pragma unroll
  for (int i = 0; i < 8; ++i) b.u[i] = *(const unsigned int*)&r[half * 16 + 2 * i];
  return b.v;
}
static __device__ __forceinline__ bf16x16 vfrag_col_gb(const __bf16* base, int n,
                                                       int half) {
  bf16x16 b;
#pragma unroll
  for (int i = 0; i < 8; ++i) {
    int kk = half * 16 + 2 * i;
    b[2 * i]     = base[(size_t)kk * D_ + n];
    b[2 * i + 1] = base[(size_t)(kk + 1) * D_ + n];
  }
  return b;
}
// A-layout fragment from un-padded LDS tile (row stride 32) — attention P
static __device__ __forceinline__ bf16x16 lds_afrag32(const __bf16* P, int m, int half) {
  union { bf16x16 v; unsigned int u[8]; } r;
#pragma unroll
  for (int i = 0; i < 8; ++i) r.u[i] = *(const unsigned int*)&P[m * 32 + a_koff(i, half)];
  return r.v;
}

// ---------------------------------------------------------------------------
// Weight prep: fp32 -> bf16 (optionally transposed to [N][K])
// ---------------------------------------------------------------------------
__global__ __launch_bounds__(256) void cvt_bf16_kernel(const float* __restrict__ src,
                                                       __bf16* __restrict__ dst) {
  size_t i = (size_t)blockIdx.x * 256 + threadIdx.x;
  dst[i] = (__bf16)src[i];
}
__global__ __launch_bounds__(256) void cvt_t_bf16_kernel(const float* __restrict__ src,
                                                         __bf16* __restrict__ dst,
                                                         int K, int N) {
  size_t i = (size_t)blockIdx.x * 256 + threadIdx.x;  // over dst [N][K]
  int n = (int)(i / K), k = (int)(i % K);
  dst[i] = (__bf16)src[(size_t)k * N + n];
}

// ---------------------------------------------------------------------------
// Embedding + sinusoidal PE: fp32 residual stream + bf16 matrix stream
// ---------------------------------------------------------------------------
__global__ __launch_bounds__(256) void embed_kernel(const int* __restrict__ tokens,
                                                    const float* __restrict__ emb,
                                                    float* __restrict__ xf,
                                                    __bf16* __restrict__ xb) {
  int idx = blockIdx.x * 256 + threadIdx.x;  // over NTOK * D
  int d   = idx & (D_ - 1);
  int row = idx >> 9;
  int s   = row & (S_ - 1);
  int t   = tokens[row];
  float e = emb[(size_t)t * D_ + d] * 22.627416998f;  // sqrt(512)
  int   i2    = d & ~1;
  float angle = (float)s * __expf(-(float)i2 * (9.2103403720f / (float)D_));
  float pe    = (d & 1) ? __cosf(angle) : __sinf(angle);
  float o     = e + pe;
  xf[idx] = o;
  xb[idx] = (__bf16)o;
}

// ---------------------------------------------------------------------------
// WMMA GEMM, all-bf16 operands, TDM-staged tiles, double buffered.
//   C[M,N] = act(alpha * A[M,K] @ W^T + bias), W given as [N][K] bf16.
// Block: 256 thr (8 waves); tile 128(M) x 64(N); Kchunk 64 -> 8 WMMA/wave/chunk.
// ---------------------------------------------------------------------------
__global__ __launch_bounds__(256) void gemm_kernel(
    const __bf16* __restrict__ A, const __bf16* __restrict__ W,
    const float* __restrict__ bias, void* __restrict__ Cout,
    int M, int N, int K, int ldc, float alpha, int relu, int obf) {
  __shared__ __bf16 As[2][128 * 68 + 8];
  __shared__ __bf16 Bs[2][64 * 68 + 8];

  const int tid  = threadIdx.x;
  const int wave = tid >> 5;
  const int lane = tid & 31;
  const int half = lane >> 4;
  const int ln   = lane & 15;

  const int m0 = blockIdx.y * 128;
  const int n0 = blockIdx.x * 64;
  const int mw = wave * 16;

  f32x8 c[4] = {{}, {}, {}, {}};

  // Prologue: kick off chunk 0 via the Tensor Data Mover (wave 0 only)
  if (tid < 32) {
    tdm_load_2d((unsigned)(size_t)&As[0][0], A + (size_t)m0 * K, 64, 128,
                (unsigned)K, (unsigned)(M - m0), (unsigned)K);
    tdm_load_2d((unsigned)(size_t)&Bs[0][0], W + (size_t)n0 * K, 64, 64,
                (unsigned)K, (unsigned)(N - n0), (unsigned)K);
  }

  for (int k0 = 0; k0 < K; k0 += 64) {
    const int cur = (k0 >> 6) & 1;
    if (tid < 32) {
      if (k0 + 64 < K) {  // prefetch next chunk, then wait for current (<=2 live)
        tdm_load_2d((unsigned)(size_t)&As[cur ^ 1][0],
                    A + (size_t)m0 * K + (k0 + 64), 64, 128,
                    (unsigned)(K - k0 - 64), (unsigned)(M - m0), (unsigned)K);
        tdm_load_2d((unsigned)(size_t)&Bs[cur ^ 1][0],
                    W + (size_t)n0 * K + (k0 + 64), 64, 64,
                    (unsigned)(K - k0 - 64), (unsigned)(N - n0), (unsigned)K);
        __builtin_amdgcn_s_wait_tensorcnt(2);
      } else {
        __builtin_amdgcn_s_wait_tensorcnt(0);
      }
    }
    __syncthreads();

    const __bf16* Ap = &As[cur][0];
    const __bf16* Bp = &Bs[cur][0];
    bf16x16 a0 = lds_frag68_a(Ap, mw + ln, 0, half);
    bf16x16 a1 = lds_frag68_a(Ap, mw + ln, 32, half);
#pragma unroll
    for (int nt = 0; nt < 4; ++nt) {
      bf16x16 b0 = lds_frag68_b(Bp, nt * 16 + ln, 0, half);
      bf16x16 b1 = lds_frag68_b(Bp, nt * 16 + ln, 32, half);
      c[nt] = wmma_bf16(a0, b0, c[nt]);
      c[nt] = wmma_bf16(a1, b1, c[nt]);
    }
    __syncthreads();
  }

  // Epilogue
#pragma unroll
  for (int nt = 0; nt < 4; ++nt) {
#pragma unroll
    for (int r = 0; r < 8; ++r) {
      int mm = m0 + mw + r + 8 * half;
      int nn = n0 + nt * 16 + ln;
      if (nn < N) {
        float v0 = c[nt][r] * alpha;
        if (bias) v0 += bias[nn];
        if (relu) v0 = fmaxf(v0, 0.0f);
        if (obf) ((__bf16*)Cout)[(size_t)mm * ldc + nn] = (__bf16)v0;
        else     ((float*)Cout)[(size_t)mm * ldc + nn]  = v0;
      }
    }
  }
}

// ---------------------------------------------------------------------------
// Causal flash-attention, bf16 q/k/v/ctx (q pre-scaled by 1/sqrt(dk)).
// Grid (S/128, H, B); one wave per 16-query tile; 8 WMMA per 32-key step.
// ---------------------------------------------------------------------------
__global__ __launch_bounds__(256) void attn_kernel(const __bf16* __restrict__ q,
                                                   const __bf16* __restrict__ k,
                                                   const __bf16* __restrict__ v,
                                                   __bf16* __restrict__ ctx) {
  __shared__ __bf16 pbuf[8][16 * 32];  // per-wave P tile

  const int wave = threadIdx.x >> 5;
  const int lane = threadIdx.x & 31;
  const int half = lane >> 4;
  const int ln   = lane & 15;

  const int b  = blockIdx.z;
  const int hh = blockIdx.y;
  const int t0 = (blockIdx.x * 8 + wave) * 16;

  const __bf16* qbase = q + ((size_t)b * S_ + t0) * D_ + hh * DK_;
  const __bf16* kbase = k + ((size_t)b * S_) * D_ + hh * DK_;
  const __bf16* vbase = v + ((size_t)b * S_) * D_ + hh * DK_;

  const bf16x16 qa0 = afrag_gb(qbase, ln, 0, half);
  const bf16x16 qa1 = afrag_gb(qbase, ln, 32, half);

  f32x8 acc0 = {}, acc1 = {}, acc2 = {}, acc3 = {};
  float mrow[8], lrow[8];
#pragma unroll
  for (int r = 0; r < 8; ++r) { mrow[r] = -1e30f; lrow[r] = 0.0f; }

  __bf16* pw = &pbuf[wave][0];

  for (int s0 = 0; s0 <= t0 + 15; s0 += 32) {
    if (s0 + 32 <= t0 + 15) {  // global_prefetch_b8 of next key block
      __builtin_prefetch(kbase + (size_t)(s0 + 32) * D_ + lane * 2, 0, 1);
    }
    f32x8 sc0 = {}, sc1 = {};
    {
      const __bf16* k0p = kbase + (size_t)s0 * D_;
      const __bf16* k1p = kbase + (size_t)(s0 + 16) * D_;
      sc0 = wmma_bf16(qa0, btfrag_gb(k0p, ln, 0, half), sc0);
      sc0 = wmma_bf16(qa1, btfrag_gb(k0p, ln, 32, half), sc0);
      sc1 = wmma_bf16(qa0, btfrag_gb(k1p, ln, 0, half), sc1);
      sc1 = wmma_bf16(qa1, btfrag_gb(k1p, ln, 32, half), sc1);
    }
#pragma unroll
    for (int r = 0; r < 8; ++r) {
      int t = t0 + r + 8 * half;
      int s = s0 + ln;
      float x0 = (s <= t) ? sc0[r] : -1e30f;
      float x1 = (s + 16 <= t) ? sc1[r] : -1e30f;
      float mx = fmaxf(x0, x1);
#pragma unroll
      for (int off = 8; off >= 1; off >>= 1) mx = fmaxf(mx, __shfl_xor(mx, off, 32));
      float mn   = fmaxf(mrow[r], mx);
      float corr = __expf(mrow[r] - mn);
      float p0   = __expf(x0 - mn);
      float p1   = __expf(x1 - mn);
      float rs   = p0 + p1;
#pragma unroll
      for (int off = 8; off >= 1; off >>= 1) rs += __shfl_xor(rs, off, 32);
      lrow[r] = lrow[r] * corr + rs;
      mrow[r] = mn;
      acc0[r] *= corr; acc1[r] *= corr; acc2[r] *= corr; acc3[r] *= corr;
      int m = r + 8 * half;
      pw[m * 32 + ln]      = (__bf16)p0;
      pw[m * 32 + 16 + ln] = (__bf16)p1;
    }
    // wave-private LDS RAW: CDNA5 split-counter wait
    asm volatile("s_wait_dscnt 0" ::: "memory");

    bf16x16 pa = lds_afrag32(pw, ln, half);
    const __bf16* vp = vbase + (size_t)s0 * D_;
    acc0 = wmma_bf16(pa, vfrag_col_gb(vp, 0  + ln, half), acc0);
    acc1 = wmma_bf16(pa, vfrag_col_gb(vp, 16 + ln, half), acc1);
    acc2 = wmma_bf16(pa, vfrag_col_gb(vp, 32 + ln, half), acc2);
    acc3 = wmma_bf16(pa, vfrag_col_gb(vp, 48 + ln, half), acc3);
  }

#pragma unroll
  for (int r = 0; r < 8; ++r) {
    int    t   = t0 + r + 8 * half;
    float  inv = 1.0f / lrow[r];
    __bf16* o  = ctx + ((size_t)b * S_ + t) * D_ + hh * DK_;
    o[0  + ln] = (__bf16)(acc0[r] * inv);
    o[16 + ln] = (__bf16)(acc1[r] * inv);
    o[32 + ln] = (__bf16)(acc2[r] * inv);
    o[48 + ln] = (__bf16)(acc3[r] * inv);
  }
}

// ---------------------------------------------------------------------------
// out = LayerNorm(a_f32 + b_bf16); writes fp32 residual + bf16 matrix copies
// ---------------------------------------------------------------------------
__global__ __launch_bounds__(256) void add_ln_kernel(const float* __restrict__ a,
                                                     const __bf16* __restrict__ bsrc,
                                                     const float* __restrict__ gamma,
                                                     const float* __restrict__ beta,
                                                     float* __restrict__ outf,
                                                     __bf16* __restrict__ outb) {
  const int wave = threadIdx.x >> 5;
  const int lane = threadIdx.x & 31;
  const int row  = blockIdx.x * 8 + wave;
  const float*  ar = a    + (size_t)row * D_;
  const __bf16* br = bsrc + (size_t)row * D_;

  float vals[16];
  float s = 0.0f;
#pragma unroll
  for (int i = 0; i < 16; ++i) {
    int d   = i * 32 + lane;
    vals[i] = ar[d] + (float)br[d];
    s += vals[i];
  }
#pragma unroll
  for (int off = 16; off >= 1; off >>= 1) s += __shfl_xor(s, off, 32);
  float mu = s * (1.0f / (float)D_);

  float vsum = 0.0f;
#pragma unroll
  for (int i = 0; i < 16; ++i) { float dv = vals[i] - mu; vsum += dv * dv; }
#pragma unroll
  for (int off = 16; off >= 1; off >>= 1) vsum += __shfl_xor(vsum, off, 32);
  float inv = rsqrtf(vsum * (1.0f / (float)D_) + 1e-6f);

  float*  of = outf + (size_t)row * D_;
  __bf16* ob = outb + (size_t)row * D_;
#pragma unroll
  for (int i = 0; i < 16; ++i) {
    int d = i * 32 + lane;
    float o = (vals[i] - mu) * inv * gamma[d] + beta[d];
    of[d] = o;
    ob[d] = (__bf16)o;
  }
}

// ---------------------------------------------------------------------------
// Host-side orchestration
// ---------------------------------------------------------------------------
extern "C" void kernel_launch(void* const* d_in, const int* in_sizes, int n_in,
                              void* d_out, int out_size, void* d_ws, size_t ws_size,
                              hipStream_t stream) {
  const int*   tokens = (const int*)d_in[0];
  const float* emb    = (const float*)d_in[1];
  const float* Wq     = (const float*)d_in[2];
  const float* Wk     = (const float*)d_in[3];
  const float* Wv     = (const float*)d_in[4];
  const float* Wo     = (const float*)d_in[5];
  const float* W1     = (const float*)d_in[6];
  const float* b1     = (const float*)d_in[7];
  const float* W2     = (const float*)d_in[8];
  const float* b2     = (const float*)d_in[9];
  const float* g1     = (const float*)d_in[10];
  const float* be1    = (const float*)d_in[11];
  const float* g2     = (const float*)d_in[12];
  const float* be2    = (const float*)d_in[13];
  float* out = (float*)d_out;

  // ---- workspace carve-up ----
  char* p = (char*)d_ws;
  auto alloc = [&](size_t bytes) -> char* {
    char* r = p; p += (bytes + 255) & ~(size_t)255; return r;
  };
  const size_t nd = (size_t)NTOK * D_;
  float*  x   = (float*)alloc(nd * 4);
  float*  h   = (float*)alloc(nd * 4);
  __bf16* xb  = (__bf16*)alloc(nd * 2);
  __bf16* hb  = (__bf16*)alloc(nd * 2);
  __bf16* tmp = (__bf16*)alloc(nd * 2);
  __bf16* big = (__bf16*)alloc(nd * 2 * 4);  // q|k|v|ctx, reused as ffn hidden
  __bf16* qb  = big;
  __bf16* kb  = qb + nd;
  __bf16* vb  = kb + nd;
  __bf16* cx  = vb + nd;
  __bf16* f1  = big;
  __bf16* wqT = (__bf16*)alloc((size_t)L_ * D_ * D_ * 2);
  __bf16* wkT = (__bf16*)alloc((size_t)L_ * D_ * D_ * 2);
  __bf16* wvT = (__bf16*)alloc((size_t)L_ * D_ * D_ * 2);
  __bf16* woC = (__bf16*)alloc((size_t)L_ * D_ * D_ * 2);
  __bf16* w1T = (__bf16*)alloc((size_t)L_ * D_ * F_ * 2);
  __bf16* w2T = (__bf16*)alloc((size_t)L_ * D_ * F_ * 2);
  __bf16* embC = (__bf16*)alloc((size_t)V_ * D_ * 2);

  const dim3 blk(256);

  // ---- weight prep: bf16, [N][K] layout ----
  for (int l = 0; l < L_; ++l) {
    const size_t oD = (size_t)l * D_ * D_;
    const size_t oF = (size_t)l * D_ * F_;
    cvt_t_bf16_kernel<<<(D_ * D_) / 256, blk, 0, stream>>>(Wq + oD, wqT + oD, D_, D_);
    cvt_t_bf16_kernel<<<(D_ * D_) / 256, blk, 0, stream>>>(Wk + oD, wkT + oD, D_, D_);
    cvt_t_bf16_kernel<<<(D_ * D_) / 256, blk, 0, stream>>>(Wv + oD, wvT + oD, D_, D_);
    cvt_bf16_kernel  <<<(D_ * D_) / 256, blk, 0, stream>>>(Wo + oD, woC + oD);
    cvt_t_bf16_kernel<<<(D_ * F_) / 256, blk, 0, stream>>>(W1 + oF, w1T + oF, D_, F_);
    cvt_t_bf16_kernel<<<(D_ * F_) / 256, blk, 0, stream>>>(W2 + oF, w2T + oF, F_, D_);
  }
  cvt_bf16_kernel<<<(V_ * D_) / 256, blk, 0, stream>>>(emb, embC);

  embed_kernel<<<(NTOK * D_) / 256, blk, 0, stream>>>(tokens, emb, x, xb);

  const float scale = 0.125f;  // 1/sqrt(64)
  const dim3 gD(D_ / 64, NTOK / 128);        // (8, 32)
  const dim3 gF(F_ / 64, NTOK / 128);        // (32, 32)
  const dim3 gV((V_ + 63) / 64, NTOK / 128); // (7, 32)
  const dim3 gA(S_ / 128, H_, B_);

  for (int l = 0; l < L_; ++l) {
    const size_t oD = (size_t)l * D_ * D_;
    const size_t oF = (size_t)l * D_ * F_;

    gemm_kernel<<<gD, blk, 0, stream>>>(xb, wqT + oD, nullptr, qb, NTOK, D_, D_, D_, scale, 0, 1);
    gemm_kernel<<<gD, blk, 0, stream>>>(xb, wkT + oD, nullptr, kb, NTOK, D_, D_, D_, 1.0f, 0, 1);
    gemm_kernel<<<gD, blk, 0, stream>>>(xb, wvT + oD, nullptr, vb, NTOK, D_, D_, D_, 1.0f, 0, 1);
    attn_kernel<<<gA, blk, 0, stream>>>(qb, kb, vb, cx);
    gemm_kernel<<<gD, blk, 0, stream>>>(cx, woC + oD, nullptr, tmp, NTOK, D_, D_, D_, 1.0f, 0, 1);
    add_ln_kernel<<<NTOK / 8, blk, 0, stream>>>(x, tmp, g1 + (size_t)l * D_, be1 + (size_t)l * D_, h, hb);
    gemm_kernel<<<gF, blk, 0, stream>>>(hb, w1T + oF, b1 + (size_t)l * F_, f1, NTOK, F_, D_, F_, 1.0f, 1, 1);
    gemm_kernel<<<gD, blk, 0, stream>>>(f1, w2T + oF, b2 + (size_t)l * D_, tmp, NTOK, D_, F_, D_, 1.0f, 0, 1);
    add_ln_kernel<<<NTOK / 8, blk, 0, stream>>>(h, tmp, g2 + (size_t)l * D_, be2 + (size_t)l * D_, x, xb);
  }

  // Tied output projection: logits = x @ emb^T (embC is [V][D] = [N][K]) -> fp32
  gemm_kernel<<<gV, blk, 0, stream>>>(xb, embC, nullptr, out, NTOK, V_, D_, V_, 1.0f, 0, 0);

  (void)in_sizes; (void)n_in; (void)out_size; (void)ws_size;
}
